// PerformerLM_block_52785148068423
// MI455X (gfx1250) — matrix operand order repaired and checked by reading, not compiled, not run on hardware
//
#include <hip/hip_runtime.h>
#include <hip/hip_bf16.h>

// ---------------- Performer LM forward, gfx1250 WMMA implementation ---------
// Heavy contractions -> v_wmma_f32_16x16x32_f16, 128x64x32 block tile,
// fragment-order LDS staging (contiguous 32B v16h fragment loads = 2x
// ds_load_b128), vectorized float4 global staging for row-contiguous AND
// column-contiguous (transposed) A operands.

typedef __attribute__((ext_vector_type(16))) _Float16 v16h;
typedef __attribute__((ext_vector_type(8)))  float    v8f;

// ---- model constants --------------------------------------------------------
#define BATCH   2
#define SEQ     8192
#define DMODEL  512
#define DEPTH   6
#define HEADS   8
#define DHEAD   64
#define INNER   512
#define NFEAT   266
#define NFP     272
#define FFDIM   2048
#define NVOCAB  5
#define ZT      (BATCH*HEADS)
#define ROWS    (BATCH*SEQ)

#define C_DN    0.35355339059327373f   // 64^-0.25
#define C_DN2   0.125f                 // dn^2
#define C_RATIO 0.06131393394849658f   // 266^-0.5
#define C_EPS   1e-4f

// ---- GEMM tile config -------------------------------------------------------
#define BM 128
#define BN 64
#define BK 32

// ============================================================================
// Generic batched GEMM:  C[z] = act( A[z] @ B[z] + bias ) * rscale + resid
// A element (m,k) at Ab[m*lda_m + k*lda_k]  (strides express transposes)
// batch base = ptr + (z/zdiv)*s?o + (z%zdiv)*s?i
//
// LDS layouts (fragment-ready, ISA 7.12.2):
//   As[strip][lane][i] : strip = row/16 (8 strips); for element (row,k):
//     lane = (row&15) + 16*((k>>3)&1), i = (k&7) + (k&16 ? 8 : 0)
//     -> per-lane v16h fragment is 32 contiguous bytes
//   Bs[col][k] (64 x 32): lane (col = lane&15) reads k = 8h..8h+7 and 16+8h..16+8h+7
//     (h = lane/16) -> per-lane v16h fragment is two 16-byte halves
// ============================================================================
__global__ __launch_bounds__(256) void k_gemm_wmma(
    const float* __restrict__ A,  long long sAo, long long sAi,
    long long lda_m, long long lda_k,
    const float* __restrict__ Bm, long long sBo, long long sBi,
    long long ldb_k, long long ldb_n,
    float* __restrict__ C,        long long sCo, long long sCi, long long ldc,
    const float* __restrict__ bias,
    const float* __restrict__ resid,
    const float* __restrict__ rscale, long long sRo, long long sRi,
    int M, int N, int K, int zdiv, int act)
{
    __shared__ alignas(32) _Float16 As[8 * 32 * 16];   // 4096 halves, 8KB
    __shared__ alignas(32) _Float16 Bs[64 * 32];       // 2048 halves, 4KB

    const int tid  = threadIdx.x;
    const int lane = tid & 31;
    const int wid  = tid >> 5;
    const int wrow = wid & 3;   // 4 waves along M (32 rows each)
    const int wcol = wid >> 2;  // 2 waves along N (32 cols each)

    const int z  = blockIdx.z;
    const int zo = z / zdiv, zi = z % zdiv;
    const float* Ab = A  + (size_t)zo * sAo + (size_t)zi * sAi;
    const float* Bb = Bm + (size_t)zo * sBo + (size_t)zi * sBi;
    float*       Cb = C  + (size_t)zo * sCo + (size_t)zi * sCi;

    const int m0 = blockIdx.y * BM;
    const int n0 = blockIdx.x * BN;

    v8f acc00 = {}, acc01 = {}, acc10 = {}, acc11 = {};

    const bool fastA  = (lda_k == 1) && ((lda_m & 3) == 0);  // row-contig A
    const bool fastAT = (lda_m == 1) && ((lda_k & 3) == 0);  // col-contig A
    const bool fastB  = (ldb_n == 1) && ((ldb_k & 3) == 0);
    const bool fullM  = (m0 + BM) <= M;
    const bool fullN  = (n0 + BN) <= N;

    const int kt = (K + BK - 1) / BK;
    for (int kk = 0; kk < kt; ++kk) {
        const int k0 = kk * BK;
        const bool fullK = (k0 + BK) <= K;

        if (kk + 1 < kt) {
            __builtin_prefetch(&Ab[(size_t)(m0 + (tid & 127)) * lda_m +
                                   (size_t)(k0 + BK) * lda_k], 0, 1);
            __builtin_prefetch(&Bb[(size_t)(k0 + BK) * ldb_k +
                                   (size_t)(n0 + (tid & 63)) * ldb_n], 0, 1);
        }

        // ---- stage A: 128x32 elements = 1024 float4 groups, 4/thread ----
        if (fastA && fullM && fullK) {
            // row-contiguous: float4 along k, single ds_store_b64
            #pragma unroll
            for (int it = 0; it < 4; ++it) {
                const int g   = tid + it * 256;
                const int row = g >> 3;
                const int kq  = (g & 7) * 4;
                const float4 f = *(const float4*)(Ab +
                    (size_t)(m0 + row) * lda_m + (size_t)(k0 + kq));
                const int laneA = (row & 15) + (((kq >> 3) & 1) << 4);
                const int ib    = (kq & 7) + ((kq & 16) ? 8 : 0);
                const int addr  = (((row >> 4) * 32 + laneA) << 4) + ib;
                union { _Float16 h[4]; unsigned long long u; } p;
                p.h[0] = (_Float16)f.x; p.h[1] = (_Float16)f.y;
                p.h[2] = (_Float16)f.z; p.h[3] = (_Float16)f.w;
                *(unsigned long long*)(&As[addr]) = p.u;
            }
        } else if (fastAT && fullM && fullK) {
            // column-contiguous (transposed view, e.g. kp^T): float4 along m
            #pragma unroll
            for (int it = 0; it < 4; ++it) {
                const int g  = tid + it * 256;
                const int r4 = (g & 31) * 4;          // row group, 4 rows
                const int k  = g >> 5;                // 0..31
                const float4 f = *(const float4*)(Ab +
                    (size_t)(m0 + r4) + (size_t)(k0 + k) * lda_k);
                const int laneHi = (((k >> 3) & 1) << 4);
                const int ib     = (k & 7) + ((k & 16) ? 8 : 0);
                const float fv[4] = {f.x, f.y, f.z, f.w};
                #pragma unroll
                for (int j = 0; j < 4; ++j) {
                    const int row   = r4 + j;
                    const int laneA = (row & 15) + laneHi;
                    As[(((row >> 4) * 32 + laneA) << 4) + ib] = (_Float16)fv[j];
                }
            }
        } else {
            #pragma unroll
            for (int it = 0; it < 4; ++it) {
                const int g   = tid + it * 256;
                const int row = g >> 3;
                const int kq  = (g & 7) * 4;
                const int gm  = m0 + row;
                float f[4];
                #pragma unroll
                for (int j = 0; j < 4; ++j) {
                    const int gk = k0 + kq + j;
                    f[j] = (gm < M && gk < K)
                         ? Ab[(size_t)gm * lda_m + (size_t)gk * lda_k] : 0.f;
                }
                const int laneA = (row & 15) + (((kq >> 3) & 1) << 4);
                const int ib    = (kq & 7) + ((kq & 16) ? 8 : 0);
                const int addr  = (((row >> 4) * 32 + laneA) << 4) + ib;
                union { _Float16 h[4]; unsigned long long u; } p;
                p.h[0] = (_Float16)f[0]; p.h[1] = (_Float16)f[1];
                p.h[2] = (_Float16)f[2]; p.h[3] = (_Float16)f[3];
                *(unsigned long long*)(&As[addr]) = p.u;
            }
        }

        // ---- stage B: 32x64 elements = 512 float4 groups, 2/thread ----
        if (fastB && fullN && fullK) {
            #pragma unroll
            for (int it = 0; it < 2; ++it) {
                const int g  = tid + it * 256;
                const int k  = g >> 4;
                const int cq = (g & 15) * 4;
                const float4 f = *(const float4*)(Bb +
                    (size_t)(k0 + k) * ldb_k + (size_t)(n0 + cq));
                Bs[(cq + 0) * 32 + k] = (_Float16)f.x;
                Bs[(cq + 1) * 32 + k] = (_Float16)f.y;
                Bs[(cq + 2) * 32 + k] = (_Float16)f.z;
                Bs[(cq + 3) * 32 + k] = (_Float16)f.w;
            }
        } else {
            #pragma unroll
            for (int it = 0; it < 2; ++it) {
                const int g  = tid + it * 256;
                const int k  = g >> 4;
                const int cq = (g & 15) * 4;
                const int gk = k0 + k;
                #pragma unroll
                for (int j = 0; j < 4; ++j) {
                    const int gn = n0 + cq + j;
                    float v = (gk < K && gn < N)
                            ? Bb[(size_t)gk * ldb_k + (size_t)gn * ldb_n] : 0.f;
                    Bs[(cq + j) * 32 + k] = (_Float16)v;
                }
            }
        }
        __syncthreads();

        // ---- fragments: A contiguous 32B v16h loads; B two 16B halves (k=8h..8h+7, 16+8h..16+8h+7) ----
        const v16h a0 = *(const v16h*)&As[(((wrow * 2 + 0) * 32 + lane) << 4)];
        const v16h a1 = *(const v16h*)&As[(((wrow * 2 + 1) * 32 + lane) << 4)];
        const int  bc = wcol * 32 + (lane & 15);
        const int  kb = (lane < 16) ? 0 : 8;
        typedef __attribute__((ext_vector_type(8))) _Float16 v8h_;
        const v8h_ b0l = *(const v8h_*)&Bs[bc * 32 + kb];
        const v8h_ b0h = *(const v8h_*)&Bs[bc * 32 + 16 + kb];
        const v8h_ b1l = *(const v8h_*)&Bs[(bc + 16) * 32 + kb];
        const v8h_ b1h = *(const v8h_*)&Bs[(bc + 16) * 32 + 16 + kb];
        const v16h b0 = __builtin_shufflevector(b0l, b0h, 0,1,2,3,4,5,6,7,8,9,10,11,12,13,14,15);
        const v16h b1 = __builtin_shufflevector(b1l, b1h, 0,1,2,3,4,5,6,7,8,9,10,11,12,13,14,15);

        acc00 = __builtin_amdgcn_wmma_f32_16x16x32_f16(
                    false, a0, false, b0, (short)0, acc00, false, false);
        acc01 = __builtin_amdgcn_wmma_f32_16x16x32_f16(
                    false, a0, false, b1, (short)0, acc01, false, false);
        acc10 = __builtin_amdgcn_wmma_f32_16x16x32_f16(
                    false, a1, false, b0, (short)0, acc10, false, false);
        acc11 = __builtin_amdgcn_wmma_f32_16x16x32_f16(
                    false, a1, false, b1, (short)0, acc11, false, false);
        __syncthreads();
    }

    // ---- epilogue: bias -> gelu(exact erf) -> row scale -> residual ----
    const float* rb = rscale ? (rscale + (size_t)zo * sRo + (size_t)zi * sRi)
                             : (const float*)nullptr;
    #pragma unroll
    for (int s = 0; s < 2; ++s) {
        #pragma unroll
        for (int j = 0; j < 2; ++j) {
            const v8f cc = (s == 0) ? (j == 0 ? acc00 : acc01)
                                    : (j == 0 ? acc10 : acc11);
            const int n = n0 + wcol * 32 + j * 16 + (lane & 15);
            if (n >= N) continue;
            #pragma unroll
            for (int r = 0; r < 8; ++r) {
                const int m = m0 + wrow * 32 + s * 16 + r
                            + ((lane < 16) ? 0 : 8);
                if (m >= M) continue;
                float v = cc[r];
                if (bias) v += bias[n];
                if (act)  v = 0.5f * v * (1.f + erff(v * 0.70710678118654752f));
                if (rb)   v *= rb[m];
                const size_t off = (size_t)m * ldc + n;
                if (resid) v += resid[off];
                Cb[off] = v;
            }
        }
    }
}

// ============================================================================
// Elementwise / reduction kernels
// ============================================================================
__global__ void k_embed(const int* __restrict__ tok,
                        const float* __restrict__ temb,
                        const float* __restrict__ pemb,
                        float* __restrict__ x, long long total)
{
    long long i = (long long)blockIdx.x * blockDim.x + threadIdx.x;
    if (i >= total) return;
    int d = (int)(i % DMODEL);
    long long row = i / DMODEL;
    int n = (int)(row % SEQ);
    int t = tok[row];
    x[i] = temb[(size_t)t * DMODEL + d] + pemb[(size_t)n * DMODEL + d];
}

__global__ __launch_bounds__(256) void k_layernorm(
    const float* __restrict__ x, const float* __restrict__ w,
    const float* __restrict__ b, float* __restrict__ y, int D)
{
    const int row = blockIdx.x;
    const float* xr = x + (size_t)row * D;
    __shared__ float ss[256], sq[256];
    float s = 0.f, q = 0.f;
    for (int d = threadIdx.x; d < D; d += 256) {
        float v = xr[d]; s += v; q += v * v;
    }
    ss[threadIdx.x] = s; sq[threadIdx.x] = q;
    __syncthreads();
    for (int off = 128; off > 0; off >>= 1) {
        if (threadIdx.x < off) {
            ss[threadIdx.x] += ss[threadIdx.x + off];
            sq[threadIdx.x] += sq[threadIdx.x + off];
        }
        __syncthreads();
    }
    const float mu  = ss[0] / D;
    const float var = sq[0] / D - mu * mu;
    const float inv = rsqrtf(var + 1e-5f);
    for (int d = threadIdx.x; d < D; d += 256)
        y[(size_t)row * D + d] = (xr[d] - mu) * inv * w[d] + b[d];
}

__global__ void k_projT(const float* __restrict__ proj, float* __restrict__ pT)
{
    int i = blockIdx.x * blockDim.x + threadIdx.x;
    if (i >= DHEAD * NFP) return;
    int d = i / NFP, m = i % NFP;
    pT[i] = (m < NFEAT) ? proj[(size_t)m * DHEAD + d] * C_DN : 0.f;
}

__global__ __launch_bounds__(256) void k_diag_rowmax(
    const float* __restrict__ xbuf,   // [BATCH,SEQ,INNER]
    const float* __restrict__ dd,     // [ZT,SEQ,NFP]
    float* __restrict__ diag, float* __restrict__ rmax)
{
    const int lane = threadIdx.x & 31;
    const int wid  = threadIdx.x >> 5;
    const long long row = (long long)blockIdx.x * 8 + wid;   // z*SEQ + n
    if (row >= (long long)ZT * SEQ) return;
    const int z = (int)(row / SEQ), n = (int)(row % SEQ);
    const int b = z / HEADS, h = z % HEADS;
    const float* xr = xbuf + (size_t)b * SEQ * INNER + (size_t)n * INNER
                           + (size_t)h * DHEAD;
    float s = 0.f;
    for (int d = lane; d < DHEAD; d += 32) { float v = xr[d]; s += v * v; }
    #pragma unroll
    for (int off = 16; off > 0; off >>= 1) s += __shfl_xor(s, off);

    const float* ddr = dd + (size_t)row * NFP;
    float mx = -1e30f;
    for (int m = lane; m < NFEAT; m += 32) mx = fmaxf(mx, ddr[m]);
    #pragma unroll
    for (int off = 16; off > 0; off >>= 1) mx = fmaxf(mx, __shfl_xor(mx, off));

    if (lane == 0) { diag[row] = 0.5f * C_DN2 * s; rmax[row] = mx; }
}

__global__ __launch_bounds__(256) void k_gmax(
    const float* __restrict__ rmax, float* __restrict__ gmax)
{
    const int z = blockIdx.x;
    __shared__ float s[256];
    float mx = -1e30f;
    for (int n = threadIdx.x; n < SEQ; n += 256)
        mx = fmaxf(mx, rmax[(size_t)z * SEQ + n]);
    s[threadIdx.x] = mx;
    __syncthreads();
    for (int off = 128; off > 0; off >>= 1) {
        if (threadIdx.x < off)
            s[threadIdx.x] = fmaxf(s[threadIdx.x], s[threadIdx.x + off]);
        __syncthreads();
    }
    if (threadIdx.x == 0) gmax[z] = s[0];
}

__global__ void k_featexp(float* __restrict__ dd,
                          const float* __restrict__ diag,
                          const float* __restrict__ rmax,
                          const float* __restrict__ gmax,
                          int is_query, long long total)
{
    long long i = (long long)blockIdx.x * blockDim.x + threadIdx.x;
    if (i >= total) return;
    const int m = (int)(i % NFP);
    const long long row = i / NFP;
    float v;
    if (m >= NFEAT) {
        v = 0.f;
    } else {
        const float mv = is_query ? rmax[row] : gmax[(int)(row / SEQ)];
        v = C_RATIO * __expf(dd[i] - diag[row] - mv) + C_EPS;
    }
    dd[i] = v;
}

__global__ __launch_bounds__(256) void k_colsum(
    const float* __restrict__ kp, float* __restrict__ ksum)
{
    const int z = blockIdx.x;
    for (int m = threadIdx.x; m < NFP; m += 256) {
        const float* p = kp + (size_t)z * SEQ * NFP + m;
        float s = 0.f;
        for (int n = 0; n < SEQ; ++n) s += p[(size_t)n * NFP];
        ksum[(size_t)z * NFP + m] = s;
    }
}

__global__ void k_dinv(const float* __restrict__ qp,
                       const float* __restrict__ ksum,
                       float* __restrict__ dinv, long long rows)
{
    long long row = (long long)blockIdx.x * blockDim.x + threadIdx.x;
    if (row >= rows) return;
    const int z = (int)(row / SEQ);
    const float* q  = qp   + (size_t)row * NFP;
    const float* ks = ksum + (size_t)z * NFP;
    float s = 0.f;
    for (int m = 0; m < NFP; ++m) s += q[m] * ks[m];
    dinv[row] = 1.f / s;
}

__global__ void k_vocab(const float* __restrict__ hln,
                        const float* __restrict__ w,
                        const float* __restrict__ b,
                        float* __restrict__ out, long long total)
{
    long long i = (long long)blockIdx.x * blockDim.x + threadIdx.x;
    if (i >= total) return;
    const int v = (int)(i % NVOCAB);
    const long long row = i / NVOCAB;
    const float* hr = hln + (size_t)row * DMODEL;
    float s = b[v];
    for (int d = 0; d < DMODEL; ++d) s += hr[d] * w[(size_t)d * NVOCAB + v];
    out[i] = s;
}

// ============================================================================
// Host-side launcher
// ============================================================================
static void gemm(hipStream_t st,
                 const float* A, long long sAo, long long sAi,
                 long long lam, long long lak,
                 const float* B, long long sBo, long long sBi,
                 long long lbk, long long lbn,
                 float* C, long long sCo, long long sCi, long long ldc,
                 const float* bias, const float* resid,
                 const float* rs, long long sRo, long long sRi,
                 int M, int N, int K, int nb, int zdiv, int act)
{
    dim3 g((N + BN - 1) / BN, (M + BM - 1) / BM, nb), blk(256);
    k_gemm_wmma<<<g, blk, 0, st>>>(A, sAo, sAi, lam, lak,
                                   B, sBo, sBi, lbk, lbn,
                                   C, sCo, sCi, ldc,
                                   bias, resid, rs, sRo, sRi,
                                   M, N, K, zdiv, act);
}

extern "C" void kernel_launch(void* const* d_in, const int* in_sizes, int n_in,
                              void* d_out, int out_size, void* d_ws, size_t ws_size,
                              hipStream_t stream)
{
    const int*   tokens   = (const int*)  d_in[0];
    const float* temb     = (const float*)d_in[1];
    const float* pemb     = (const float*)d_in[2];
    const float* proj     = (const float*)d_in[3];
    const float* ln1_w    = (const float*)d_in[4];
    const float* ln1_b    = (const float*)d_in[5];
    const float* qkv_w    = (const float*)d_in[6];
    const float* qkv_b    = (const float*)d_in[7];
    const float* ao_w     = (const float*)d_in[8];
    const float* ao_b     = (const float*)d_in[9];
    const float* ln2_w    = (const float*)d_in[10];
    const float* ln2_b    = (const float*)d_in[11];
    const float* ff1_w    = (const float*)d_in[12];
    const float* ff1_b    = (const float*)d_in[13];
    const float* ff2_w    = (const float*)d_in[14];
    const float* ff2_b    = (const float*)d_in[15];
    const float* lnf_w    = (const float*)d_in[16];
    const float* lnf_b    = (const float*)d_in[17];
    const float* out_w    = (const float*)d_in[18];
    const float* out_b    = (const float*)d_in[19];
    float* out = (float*)d_out;

    // ---- workspace layout (floats) ----
    float* W = (float*)d_ws;
    size_t o = 0;
    float* x    = W + o; o += (size_t)ROWS * DMODEL;
    float* hln  = W + o; o += (size_t)ROWS * DMODEL;
    float* qb   = W + o; o += (size_t)ROWS * INNER;
    float* kb   = W + o; o += (size_t)ROWS * INNER;
    float* vb   = W + o; o += (size_t)ROWS * INNER;
    float* ob   = W + o; o += (size_t)ROWS * INNER;
    float* qp   = W + o; o += (size_t)ZT * SEQ * NFP;
    float* kp   = W + o; o += (size_t)ZT * SEQ * NFP;
    float* ffb  = W + o; o += (size_t)ROWS * FFDIM;
    float* ctx  = W + o; o += (size_t)ZT * NFP * DHEAD;
    float* ksum = W + o; o += (size_t)ZT * NFP;
    float* diag = W + o; o += (size_t)ZT * SEQ;
    float* rmax = W + o; o += (size_t)ZT * SEQ;
    float* gmx  = W + o; o += (size_t)ZT;
    float* dinv = W + o; o += (size_t)ZT * SEQ;
    float* pT   = W + o; o += (size_t)DHEAD * NFP;

    const long long EL = (long long)ROWS * DMODEL;
    k_embed<<<(unsigned)((EL + 255) / 256), 256, 0, stream>>>(tokens, temb, pemb, x, EL);

    const long long FEAT_TOT = (long long)ZT * SEQ * NFP;

    for (int i = 0; i < DEPTH; ++i) {
        // ---- LN1 ----
        k_layernorm<<<ROWS, 256, 0, stream>>>(x, ln1_w + i * DMODEL,
                                              ln1_b + i * DMODEL, hln, DMODEL);
        // ---- QKV (3 GEMMs) ----
        float* qkvbuf[3] = {qb, kb, vb};
        for (int t = 0; t < 3; ++t) {
            const float* wp = qkv_w + ((size_t)(i * 3 + t)) * DMODEL * INNER;
            const float* bp = qkv_b + (size_t)(i * 3 + t) * INNER;
            gemm(stream, hln, 0, 0, DMODEL, 1,
                 wp, 0, 0, INNER, 1,
                 qkvbuf[t], 0, 0, INNER,
                 bp, nullptr, nullptr, 0, 0,
                 ROWS, INNER, DMODEL, 1, 1, 0);
        }
        // ---- projT (fold dn) ----
        k_projT<<<(DHEAD * NFP + 255) / 256, 256, 0, stream>>>(
            proj + (size_t)i * NFEAT * DHEAD, pT);

        // ---- dd = x_head @ projT  (batched over 16 (b,h)) ----
        gemm(stream, qb, (long long)SEQ * INNER, DHEAD, INNER, 1,
             pT, 0, 0, NFP, 1,
             qp, (long long)HEADS * SEQ * NFP, (long long)SEQ * NFP, NFP,
             nullptr, nullptr, nullptr, 0, 0,
             SEQ, NFP, DHEAD, ZT, HEADS, 0);
        gemm(stream, kb, (long long)SEQ * INNER, DHEAD, INNER, 1,
             pT, 0, 0, NFP, 1,
             kp, (long long)HEADS * SEQ * NFP, (long long)SEQ * NFP, NFP,
             nullptr, nullptr, nullptr, 0, 0,
             SEQ, NFP, DHEAD, ZT, HEADS, 0);

        // ---- FAVOR+ nonlinearity ----
        k_diag_rowmax<<<ZT * SEQ / 8, 256, 0, stream>>>(qb, qp, diag, rmax);
        k_featexp<<<(unsigned)((FEAT_TOT + 255) / 256), 256, 0, stream>>>(
            qp, diag, rmax, nullptr, 1, FEAT_TOT);
        k_diag_rowmax<<<ZT * SEQ / 8, 256, 0, stream>>>(kb, kp, diag, rmax);
        k_gmax<<<ZT, 256, 0, stream>>>(rmax, gmx);
        k_featexp<<<(unsigned)((FEAT_TOT + 255) / 256), 256, 0, stream>>>(
            kp, diag, rmax, gmx, 0, FEAT_TOT);

        // ---- linear attention ----
        k_colsum<<<ZT, 256, 0, stream>>>(kp, ksum);
        k_dinv<<<(ZT * SEQ + 255) / 256, 256, 0, stream>>>(
            qp, ksum, dinv, (long long)ZT * SEQ);

        // ctx[z] = kp[z]^T @ v[z]   (M=NFP, K=SEQ, N=DHEAD; A col-contiguous)
        gemm(stream, kp, (long long)HEADS * SEQ * NFP, (long long)SEQ * NFP, 1, NFP,
             vb, (long long)SEQ * INNER, DHEAD, INNER, 1,
             ctx, (long long)HEADS * NFP * DHEAD, (long long)NFP * DHEAD, DHEAD,
             nullptr, nullptr, nullptr, 0, 0,
             NFP, DHEAD, SEQ, ZT, HEADS, 0);

        // o[z] = (qp[z] @ ctx[z]) * dinv  -> written to [b,n,h*64+e]
        gemm(stream, qp, (long long)HEADS * SEQ * NFP, (long long)SEQ * NFP, NFP, 1,
             ctx, (long long)HEADS * NFP * DHEAD, (long long)NFP * DHEAD, DHEAD, 1,
             ob, (long long)SEQ * INNER, DHEAD, INNER,
             nullptr, nullptr, dinv, (long long)HEADS * SEQ, SEQ,
             SEQ, DHEAD, NFP, ZT, HEADS, 0);

        // ---- attention output projection + residual ----
        gemm(stream, ob, 0, 0, INNER, 1,
             ao_w + (size_t)i * INNER * DMODEL, 0, 0, DMODEL, 1,
             x, 0, 0, DMODEL,
             ao_b + (size_t)i * DMODEL, x, nullptr, 0, 0,
             ROWS, DMODEL, INNER, 1, 1, 0);

        // ---- FFN ----
        k_layernorm<<<ROWS, 256, 0, stream>>>(x, ln2_w + i * DMODEL,
                                              ln2_b + i * DMODEL, hln, DMODEL);
        gemm(stream, hln, 0, 0, DMODEL, 1,
             ff1_w + (size_t)i * DMODEL * FFDIM, 0, 0, FFDIM, 1,
             ffb, 0, 0, FFDIM,
             ff1_b + (size_t)i * FFDIM, nullptr, nullptr, 0, 0,
             ROWS, FFDIM, DMODEL, 1, 1, 1 /*gelu*/);
        gemm(stream, ffb, 0, 0, FFDIM, 1,
             ff2_w + (size_t)i * FFDIM * DMODEL, 0, 0, DMODEL, 1,
             x, 0, 0, DMODEL,
             ff2_b + (size_t)i * DMODEL, x, nullptr, 0, 0,
             ROWS, DMODEL, FFDIM, 1, 1, 0);
    }

    // ---- final LN + vocab projection ----
    k_layernorm<<<ROWS, 256, 0, stream>>>(x, lnf_w, lnf_b, hln, DMODEL);
    const long long VT = (long long)ROWS * NVOCAB;
    k_vocab<<<(unsigned)((VT + 255) / 256), 256, 0, stream>>>(hln, out_w, out_b, out, VT);
}
